// MultiHeadAttention_54193897341247
// MI455X (gfx1250) — matrix operand order repaired and checked by reading, not compiled, not run on hardware
//
#include <hip/hip_runtime.h>
#include <hip/hip_bf16.h>
#include <math.h>

typedef __attribute__((ext_vector_type(16))) _Float16 v16h;
typedef __attribute__((ext_vector_type(8)))  _Float16 v8h;
typedef __attribute__((ext_vector_type(4)))  _Float16 v4h;
typedef __attribute__((ext_vector_type(4)))  float    v4f;
typedef __attribute__((ext_vector_type(8)))  float    v8f;

#define E_  1024
#define S_  64
#define H_  16
#define HD_ 64

__device__ __forceinline__ v16h cat8(v8h lo, v8h hi) {
    return __builtin_shufflevector(lo, hi, 0,1,2,3,4,5,6,7,8,9,10,11,12,13,14,15);
}

// A frag (16x32 f16): lane<16 -> M=lane, K {k0..k0+7, k0+16..k0+23};
//                     lane>=16 -> M=lane-16, K {k0+8..15, k0+24..31}.
// row pointer `pa` must already include the (hi?8:0) K shift. 16B-aligned.
__device__ __forceinline__ v16h load_afrag(const _Float16* pa) {
    v8h lo = *(const v8h*)(pa);
    v8h hi = *(const v8h*)(pa + 16);
    return cat8(lo, hi);
}

// ---------------- fp32 -> fp16 conversion (vectorized) ----------------
__global__ void cvt_f32_f16(const float* __restrict__ s, _Float16* __restrict__ d, long n4) {
    long i = (long)blockIdx.x * blockDim.x + threadIdx.x;
    long stride = (long)gridDim.x * blockDim.x;
    for (; i < n4; i += stride) {
        v4f v = *(const v4f*)(s + i * 4);
        v4h h; h.x = (_Float16)v.x; h.y = (_Float16)v.y; h.z = (_Float16)v.z; h.w = (_Float16)v.w;
        *(v4h*)(d + i * 4) = h;
    }
}

// ---------------- generic WMMA GEMM ----------------
// C(MxN) = A(MxK,f16,row-major) @ B(KxN,f16,row-major) [+ bias(N,f32)]
// Block: 8 waves, tile 128 x (16*NT). Wave w -> rows [128*bx + 16w, +16).
// B frag (32x16 f16): lane -> N=lane&15, K same table order as A frag (strided gather).
// TR64: permute stored column index within 64x64 blocks (col -> (col&63)*64 + col/64),
//       used to produce the smolgen bias transposed per head for fast reads.
template <int NT, bool OUT_HALF, bool HAS_BIAS, bool TR64 = false>
__global__ __launch_bounds__(256) void gemm_wmma(
    const _Float16* __restrict__ A, const _Float16* __restrict__ Bm,
    const float* __restrict__ bias, void* __restrict__ Cp,
    int M, int N, int K)
{
    const int wave = threadIdx.x >> 5;
    const int lane = threadIdx.x & 31;
    const int mlo  = lane & 15;
    const int hi   = (lane >= 16) ? 1 : 0;
    const int row0 = blockIdx.x * 128 + wave * 16;
    const int n0   = blockIdx.y * (NT * 16);

    v8f acc[NT] = {};
    const _Float16* arow = A + (long)(row0 + mlo) * K + (hi ? 8 : 0);
    const _Float16* brow = Bm + (long)(hi ? 8 : 0) * N + n0 + mlo;

    for (int k0 = 0; k0 < K; k0 += 32) {
        const v16h a = load_afrag(arow + k0);
        const _Float16* pb = brow + (long)k0 * N;
        #pragma unroll
        for (int t = 0; t < NT; ++t) {
            v16h b;
            #pragma unroll
            for (int j = 0; j < 16; ++j) b[j] = pb[(long)((j < 8) ? j : j + 8) * N + t * 16];
            acc[t] = __builtin_amdgcn_wmma_f32_16x16x32_f16(
                false, a, false, b, (short)0, acc[t], false, false);
        }
    }

    #pragma unroll
    for (int t = 0; t < NT; ++t) {
        const int col = n0 + t * 16 + mlo;
        const float bv = HAS_BIAS ? bias[col] : 0.0f;
        const int colw = TR64 ? (((col & 63) << 6) | (col >> 6)) : col;
        #pragma unroll
        for (int i = 0; i < 8; ++i) {
            const int row = row0 + i + hi * 8;
            const float v = acc[t][i] + bv;
            if constexpr (OUT_HALF)
                ((_Float16*)Cp)[(long)row * N + colw] = (_Float16)v;
            else
                ((float*)Cp)[(long)row * N + colw] = v;
        }
    }
}

// ---------------- SiLU + LayerNorm over rows ----------------
__global__ void silu_ln(const float* __restrict__ in, const float* __restrict__ g,
                        const float* __restrict__ be, _Float16* __restrict__ out,
                        int rows, int cols)
{
    int r = blockIdx.x * blockDim.x + threadIdx.x;
    if (r >= rows) return;
    const float* p = in + (long)r * cols;
    float sum = 0.f, sumsq = 0.f;
    for (int i = 0; i < cols; ++i) {
        float v = p[i];
        float s = v / (1.0f + expf(-v));     // silu
        sum += s; sumsq += s * s;
    }
    float mean = sum / cols;
    float var  = sumsq / cols - mean * mean;
    float inv  = rsqrtf(var + 1e-3f);
    _Float16* o = out + (long)r * cols;
    for (int i = 0; i < cols; ++i) {
        float v = p[i];
        float s = v / (1.0f + expf(-v));
        o[i] = (_Float16)((s - mean) * inv * g[i] + be[i]);
    }
}

// ---------------- fused attention per (b,h) ----------------
// logits = (Q K^T)/8 + biasT ; softmax ; O = P V   (all 64x64 per head)
// biash is stored TRANSPOSED per head: biash[bh][k][q]  (TR64 GEMM epilogue)
__global__ __launch_bounds__(128) void attn_kernel(
    const _Float16* __restrict__ Qh, const _Float16* __restrict__ Kh,
    const _Float16* __restrict__ Vh, const _Float16* __restrict__ biash,
    _Float16* __restrict__ Oh)
{
    __shared__ _Float16 lds_kt[HD_ * S_];   // K transposed: [hd][s]
    __shared__ float    lds_l[S_ * S_];
    __shared__ _Float16 lds_p[S_ * S_];

    const int bh = blockIdx.x;
    const int b = bh >> 4, h = bh & 15;
    const long base = ((long)b * S_) * E_ + h * HD_;   // (b*S + s)*E + h*HD

    // stage K tile (64x64) into LDS TRANSPOSED: lds_kt[hd*64+s] = K[s][hd].
    for (int idx = threadIdx.x * 8; idx < S_ * HD_; idx += 128 * 8) {
        const int s = idx >> 6, c = idx & 63;
        v8h kv = *(const v8h*)(Kh + base + (long)s * E_ + c);
        #pragma unroll
        for (int j = 0; j < 8; ++j) lds_kt[(c + j) * S_ + s] = kv[j];
    }
    __syncthreads();

    const int wave = threadIdx.x >> 5;
    const int lane = threadIdx.x & 31;
    const int mlo  = lane & 15;
    const int hi   = (lane >= 16) ? 1 : 0;
    const int row0 = wave * 16;

    // ---- logits = Q @ K^T ----
    v8f acc[4] = {};
    #pragma unroll
    for (int k0 = 0; k0 < HD_; k0 += 32) {
        const v16h a = load_afrag(Qh + base + (long)(row0 + mlo) * E_ + k0 + (hi ? 8 : 0));
        const _Float16* pb = lds_kt + (k0 + (hi ? 8 : 0)) * S_ + mlo;   // col n=s, k=hd in A-frag order
        #pragma unroll
        for (int t = 0; t < 4; ++t) {
            v16h bfrag;
            #pragma unroll
            for (int j = 0; j < 16; ++j) bfrag[j] = pb[((j < 8) ? j : j + 8) * S_ + t * 16];
            acc[t] = __builtin_amdgcn_wmma_f32_16x16x32_f16(
                false, a, false, bfrag, (short)0, acc[t], false, false);
        }
    }

    const float scale = 0.125f;   // HD^-0.5
    const _Float16* bbT = biash + (long)bh * (S_ * S_);   // [k][q] layout
    #pragma unroll
    for (int t = 0; t < 4; ++t) {
        const int col = t * 16 + mlo;                     // k index
        // 8 consecutive q values for this lane: one 16B load
        const v8h bvh = *(const v8h*)(bbT + col * S_ + row0 + hi * 8);
        #pragma unroll
        for (int i = 0; i < 8; ++i) {
            const int row = row0 + i + hi * 8;            // q index
            lds_l[row * S_ + col] = acc[t][i] * scale + (float)bvh[i];
        }
    }
    __syncthreads();

    // ---- softmax (one thread per row) ----
    if (threadIdx.x < S_) {
        const int r = threadIdx.x;
        float mx = -1e30f;
        for (int j = 0; j < S_; ++j) mx = fmaxf(mx, lds_l[r * S_ + j]);
        float sum = 0.f;
        for (int j = 0; j < S_; ++j) {
            float e = expf(lds_l[r * S_ + j] - mx);
            lds_l[r * S_ + j] = e; sum += e;
        }
        float inv = 1.0f / sum;
        for (int j = 0; j < S_; ++j)
            lds_p[r * S_ + j] = (_Float16)(lds_l[r * S_ + j] * inv);
    }
    __syncthreads();

    // ---- O = P @ V ----
    v8f oc[4] = {};
    #pragma unroll
    for (int k0 = 0; k0 < S_; k0 += 32) {
        const v16h a = load_afrag(lds_p + (row0 + mlo) * S_ + k0 + (hi ? 8 : 0));
        const _Float16* pb = Vh + base + (long)(k0 + (hi ? 8 : 0)) * E_ + mlo;  // col n=hd, k=s in A-frag order
        #pragma unroll
        for (int t = 0; t < 4; ++t) {
            v16h bfrag;
            #pragma unroll
            for (int j = 0; j < 16; ++j) bfrag[j] = pb[(long)((j < 8) ? j : j + 8) * E_ + t * 16];
            oc[t] = __builtin_amdgcn_wmma_f32_16x16x32_f16(
                false, a, false, bfrag, (short)0, oc[t], false, false);
        }
    }
    #pragma unroll
    for (int t = 0; t < 4; ++t) {
        const int col = t * 16 + mlo;
        #pragma unroll
        for (int i = 0; i < 8; ++i) {
            const int row = row0 + i + hi * 8;
            Oh[base + (long)row * E_ + col] = (_Float16)oc[t][i];
        }
    }
}

// ---------------- launcher ----------------
extern "C" void kernel_launch(void* const* d_in, const int* in_sizes, int n_in,
                              void* d_out, int out_size, void* d_ws, size_t ws_size,
                              hipStream_t stream)
{
    const float* x   = (const float*)d_in[0];
    const float* Wq  = (const float*)d_in[1];  const float* bq  = (const float*)d_in[2];
    const float* Wk  = (const float*)d_in[3];  const float* bk  = (const float*)d_in[4];
    const float* Wv  = (const float*)d_in[5];  const float* bv  = (const float*)d_in[6];
    const float* Wo  = (const float*)d_in[7];  const float* bo  = (const float*)d_in[8];
    const float* Wc  = (const float*)d_in[9];
    const float* Wd1 = (const float*)d_in[10]; const float* bd1 = (const float*)d_in[11];
    const float* g1  = (const float*)d_in[12]; const float* be1 = (const float*)d_in[13];
    const float* Wd2 = (const float*)d_in[14]; const float* bd2 = (const float*)d_in[15];
    const float* g2  = (const float*)d_in[16]; const float* be2 = (const float*)d_in[17];
    const float* Wg  = (const float*)d_in[18];

    const long BS = 1024L * S_;           // 65536 tokens
    char* ws = (char*)d_ws;
    size_t off = 0;
    auto alloc = [&](size_t bytes) -> void* {
        off = (off + 255) & ~(size_t)255;
        void* p = ws + off; off += bytes; return p;
    };

    _Float16* xh   = (_Float16*)alloc(BS * E_ * 2);
    _Float16* Qh   = (_Float16*)alloc(BS * E_ * 2);
    _Float16* Khb  = (_Float16*)alloc(BS * E_ * 2);
    _Float16* Vhb  = (_Float16*)alloc(BS * E_ * 2);
    _Float16* Oh   = (_Float16*)alloc(BS * E_ * 2);
    _Float16* Wqh  = (_Float16*)alloc((size_t)E_ * E_ * 2);
    _Float16* Wkh  = (_Float16*)alloc((size_t)E_ * E_ * 2);
    _Float16* Wvh  = (_Float16*)alloc((size_t)E_ * E_ * 2);
    _Float16* Woh  = (_Float16*)alloc((size_t)E_ * E_ * 2);
    _Float16* Wch  = (_Float16*)alloc((size_t)E_ * 32 * 2);
    _Float16* Wd1h = (_Float16*)alloc((size_t)2048 * 256 * 2);
    _Float16* Wd2h = (_Float16*)alloc((size_t)256 * 4096 * 2);
    _Float16* Wgh  = (_Float16*)alloc((size_t)256 * 4096 * 2);
    _Float16* comp = (_Float16*)alloc(BS * 32 * 2);          // (65536,32)=(1024,2048)
    float*    hidp = (float*)   alloc((size_t)1024 * 256 * 4);
    _Float16* hidh = (_Float16*)alloc((size_t)1024 * 256 * 2);
    float*    genp = (float*)   alloc((size_t)1024 * 4096 * 4);
    _Float16* genh = (_Float16*)alloc((size_t)1024 * 4096 * 2);  // = (16384,256)
    _Float16* bsh  = (_Float16*)alloc((size_t)16384 * 4096 * 2); // smolgen bias (B,H,64,64) TRANSPOSED [k][q]

    // --- conversions to f16 (n/4 vector elems each) ---
    cvt_f32_f16<<<4096, 256, 0, stream>>>(x,   xh,   BS * E_ / 4);
    cvt_f32_f16<<<1024, 256, 0, stream>>>(Wq,  Wqh,  (long)E_ * E_ / 4);
    cvt_f32_f16<<<1024, 256, 0, stream>>>(Wk,  Wkh,  (long)E_ * E_ / 4);
    cvt_f32_f16<<<1024, 256, 0, stream>>>(Wv,  Wvh,  (long)E_ * E_ / 4);
    cvt_f32_f16<<<1024, 256, 0, stream>>>(Wo,  Woh,  (long)E_ * E_ / 4);
    cvt_f32_f16<<<32,   256, 0, stream>>>(Wc,  Wch,  (long)E_ * 32 / 4);
    cvt_f32_f16<<<512,  256, 0, stream>>>(Wd1, Wd1h, 2048L * 256 / 4);
    cvt_f32_f16<<<1024, 256, 0, stream>>>(Wd2, Wd2h, 256L * 4096 / 4);
    cvt_f32_f16<<<1024, 256, 0, stream>>>(Wg,  Wgh,  256L * 4096 / 4);

    // --- Q/K/V projections: (65536,1024)@(1024,1024)+bias -> f16 ---
    dim3 gp(512, 16);
    gemm_wmma<4, true, true><<<gp, 256, 0, stream>>>(xh, Wqh, bq, Qh,  65536, E_, E_);
    gemm_wmma<4, true, true><<<gp, 256, 0, stream>>>(xh, Wkh, bk, Khb, 65536, E_, E_);
    gemm_wmma<4, true, true><<<gp, 256, 0, stream>>>(xh, Wvh, bv, Vhb, 65536, E_, E_);

    // --- smolgen chain ---
    gemm_wmma<2, true, false><<<dim3(512, 1), 256, 0, stream>>>(xh, Wch, nullptr, comp, 65536, 32, E_);
    gemm_wmma<4, false, true><<<dim3(8, 4),   256, 0, stream>>>(comp, Wd1h, bd1, hidp, 1024, 256, 2048);
    silu_ln<<<4, 256, 0, stream>>>(hidp, g1, be1, hidh, 1024, 256);
    gemm_wmma<4, false, true><<<dim3(8, 64),  256, 0, stream>>>(hidh, Wd2h, bd2, genp, 1024, 4096, 256);
    silu_ln<<<4, 256, 0, stream>>>(genp, g2, be2, genh, 1024, 4096);
    // bias GEMM with TR64 epilogue: bsh[bh][k][q]
    gemm_wmma<4, true, false, true><<<dim3(128, 64), 256, 0, stream>>>(genh, Wgh, nullptr, bsh, 16384, 4096, 256);

    // --- fused attention (one block per (b,h)) ---
    attn_kernel<<<16384, 128, 0, stream>>>(Qh, Khb, Vhb, bsh, Oh);

    // --- output projection: (65536,1024)@(1024,1024)+bo -> f32 d_out ---
    gemm_wmma<4, false, true><<<gp, 256, 0, stream>>>(Oh, Woh, bo, d_out, 65536, E_, E_);

    (void)in_sizes; (void)n_in; (void)out_size; (void)ws_size;
}
